// Attention_10402410790972
// MI455X (gfx1250) — matrix operand order repaired and checked by reading, not compiled, not run on hardware
//
#include <hip/hip_runtime.h>

typedef __bf16 bf16;
typedef __attribute__((ext_vector_type(16))) __bf16 v16bf;
typedef __attribute__((ext_vector_type(8)))  float  v8f;
typedef __attribute__((ext_vector_type(4)))  float  v4f;
typedef __attribute__((ext_vector_type(4)))  __bf16 v4bf;

static __device__ inline v8f wmma_bf16(v16bf a, v16bf b, v8f c) {
  // D = A(16x32) * B(32x16) + C, f32 accumulate
  return __builtin_amdgcn_wmma_f32_16x16x32_bf16(false, a, false, b, (short)0, c, false, false);
}

// CDNA5 async global->LDS copy, 128 bits per lane. Tracked by ASYNCcnt.
static __device__ inline void async_copy_g2l_b128(const void* gsrc, void* ldst) {
  unsigned lofs = (unsigned)(size_t)ldst;          // LDS addr = addr[31:0]
  unsigned long long g = (unsigned long long)(size_t)gsrc;
  asm volatile("global_load_async_to_lds_b128 %0, %1, off"
               :: "v"(lofs), "v"(g) : "memory");
}
static __device__ inline void wait_async() {
  asm volatile("s_wait_asynccnt 0" ::: "memory");
}

// A fragment 16x32 (MxK) from row-major [m][k] storage, ld in elements.
static __device__ inline v16bf frag_a(const bf16* base, int ld) {
  int lane = threadIdx.x & 31;
  const bf16* row = base + (lane & 15) * ld + ((lane >> 4) ? 16 : 0);
  v16bf f;
#pragma unroll
  for (int t = 0; t < 8; ++t) f[t] = row[t];
#pragma unroll
  for (int t = 0; t < 8; ++t) f[8 + t] = row[8 + t];
  return f;
}

// Same but rows m>=8 forced to zero (M=8-in-16 padded GEMMs; avoids OOB reads)
static __device__ inline v16bf frag_a_rows8(const bf16* base, int ld) {
  int lane = threadIdx.x & 31;
  int m = lane & 15;
  v16bf f;
  if (m < 8) {
    const bf16* row = base + m * ld + ((lane >> 4) ? 16 : 0);
#pragma unroll
    for (int t = 0; t < 8; ++t) f[t] = row[t];
#pragma unroll
    for (int t = 0; t < 8; ++t) f[8 + t] = row[8 + t];
  } else {
#pragma unroll
    for (int t = 0; t < 16; ++t) f[t] = (bf16)0.0f;
  }
  return f;
}

// B fragment 32x16 (KxN) from row-major B[k][n]
static __device__ inline v16bf frag_b(const bf16* base, int ld) {
  int lane = threadIdx.x & 31;
  const bf16* p = base + (lane & 15) + ((lane >> 4) ? 16 : 0) * ld;
  v16bf f;
#pragma unroll
  for (int t = 0; t < 16; ++t) f[t] = p[t * ld];
  return f;
}

// B fragment from transposed storage Bt[n][k] (contiguous K per lane)
static __device__ inline v16bf frag_bt(const bf16* base, int ld) {
  int lane = threadIdx.x & 31;
  const bf16* row = base + (lane & 15) * ld + ((lane >> 4) ? 16 : 0);
  v16bf f;
#pragma unroll
  for (int t = 0; t < 16; ++t) f[t] = row[t];
  return f;
}

__global__ void f32_to_bf16_v4(const v4f* __restrict__ in, v4bf* __restrict__ out, int n4) {
  int i = blockIdx.x * 256 + threadIdx.x;
  if (i < n4) {
    v4f x = in[i];
    v4bf y;
    y[0] = (bf16)x[0]; y[1] = (bf16)x[1]; y[2] = (bf16)x[2]; y[3] = (bf16)x[3];
    out[i] = y;
  }
}

// Generic GEMM: C[M,N] = A[M,K](bf16) * B[K,N](bf16, col offset bcol) + bias
// Block 256 threads = 8 waves; block tile M=32 x N=64, k-block 64.
// Staging via CDNA5 async global->LDS b128 copies; 2 WMMA per wave per stage.
template <bool OUTF32>
__global__ __launch_bounds__(256) void gemm_bf16_kernel(
    const bf16* __restrict__ A, int lda,
    const bf16* __restrict__ B, int ldb, int bcol,
    const float* __restrict__ bias,
    void* __restrict__ outp, int ldo,
    int M, int N, int K) {
  __shared__ bf16 As[32 * 64];   // [m][k]
  __shared__ bf16 Bs[64 * 64];   // [k][n]
  int m0 = blockIdx.y * 32, n0 = blockIdx.x * 64;
  int tid = threadIdx.x, wid = tid >> 5;
  int wm = wid >> 2, wn = wid & 3;
  // staging coordinates (128-bit granules)
  int am = tid >> 3, ak = (tid & 7) * 8;    // A: 32 rows x 64 k, 8 bf16 per thread
  int bk = tid >> 2, bn = (tid & 3) * 16;   // B: 64 k   x 64 n, 16 bf16 per thread
  v8f acc = {};
  for (int k0 = 0; k0 < K; k0 += 64) {
    async_copy_g2l_b128(A + (m0 + am) * lda + k0 + ak, &As[am * 64 + ak]);
    const bf16* bsrc = B + (k0 + bk) * ldb + bcol + n0 + bn;
    async_copy_g2l_b128(bsrc, &Bs[bk * 64 + bn]);
    async_copy_g2l_b128(bsrc + 8, &Bs[bk * 64 + bn + 8]);
    wait_async();
    __syncthreads();
    v16bf a0 = frag_a(As + wm * 16 * 64, 64);
    v16bf a1 = frag_a(As + wm * 16 * 64 + 32, 64);
    v16bf b0 = frag_b(Bs + wn * 16, 64);
    v16bf b1 = frag_b(Bs + 32 * 64 + wn * 16, 64);
    acc = wmma_bf16(a0, b0, acc);
    acc = wmma_bf16(a1, b1, acc);
    __syncthreads();
  }
  int lane = tid & 31;
  int ncol = n0 + wn * 16 + (lane & 15);
  int mrow = m0 + wm * 16 + ((lane >> 4) ? 8 : 0);
  float bv = bias[bcol + ncol];
#pragma unroll
  for (int r = 0; r < 8; ++r) {
    float vv = acc[r] + bv;
    if constexpr (OUTF32) ((float*)outp)[(mrow + r) * ldo + ncol] = vv;
    else                  ((bf16*)outp)[(mrow + r) * ldo + ncol] = (bf16)vv;
  }
}

// qe[m, h*64+c] = sum_d q[m, h*64+d] * W_e[c, h*64+d]   (per-head 64x64, M-tiles of 16)
__global__ __launch_bounds__(32) void qe_kernel(const bf16* __restrict__ q,
                                                const bf16* __restrict__ We,
                                                bf16* __restrict__ qe) {
  int m0 = blockIdx.x * 16;
  int h  = blockIdx.y;
  const bf16* Abase = q + m0 * 512 + h * 64;
#pragma unroll
  for (int nt = 0; nt < 4; ++nt) {
    v8f acc = {};
#pragma unroll
    for (int ks = 0; ks < 2; ++ks) {
      v16bf a = frag_a(Abase + ks * 32, 512);
      // Bt[n=c][k=d] = W_e[c*512 + h*64 + d] -> contiguous in d
      v16bf b = frag_bt(We + (nt * 16) * 512 + h * 64 + ks * 32, 512);
      acc = wmma_bf16(a, b, acc);
    }
    int lane = threadIdx.x & 31;
    int c = nt * 16 + (lane & 15);
    int mb = (lane >> 4) ? 8 : 0;
#pragma unroll
    for (int r = 0; r < 8; ++r)
      qe[(m0 + mb + r) * 512 + h * 64 + c] = (bf16)acc[r];
  }
}

// sbias[b,h,i,j] = sum_c qe[b,i,h,c] * edges[b,c,i,j]
__global__ __launch_bounds__(128) void sbias_kernel(const bf16* __restrict__ qe,
                                                    const bf16* __restrict__ edges,
                                                    float* __restrict__ sbias) {
  int i = blockIdx.x, b = blockIdx.y;
  int wid = threadIdx.x >> 5;
  const bf16* Abase = qe + (b * 320 + i) * 512;                 // [h][c], ld 64
  const bf16* Ebase = edges + b * 64 * 102400 + i * 320;        // B[k=c][n=j], ld 102400
  v16bf a0 = frag_a_rows8(Abase, 64);
  v16bf a1 = frag_a_rows8(Abase + 32, 64);
  for (int t = 0; t < 5; ++t) {
    int j0 = (wid * 5 + t) * 16;
    v8f acc = {};
    acc = wmma_bf16(a0, frag_b(Ebase + j0, 102400), acc);
    acc = wmma_bf16(a1, frag_b(Ebase + 32 * 102400 + j0, 102400), acc);
    int lane = threadIdx.x & 31;
    if (lane < 16) {
      int j = j0 + lane;
#pragma unroll
      for (int r = 0; r < 8; ++r)
        sbias[((b * 8 + r) * 320 + i) * 320 + j] = acc[r];
    }
  }
}

// Fused attention: per (b,h,i-tile of 16): sim = q@k^T + sbias, softmax, ctx = attn@v
__global__ __launch_bounds__(32) void attn_kernel(
    const bf16* __restrict__ q, const bf16* __restrict__ k,
    const bf16* __restrict__ v, const float* __restrict__ sbias,
    bf16* __restrict__ attn_out, float* __restrict__ ctx) {
  __shared__ float simbuf[16 * 320];
  __shared__ bf16  attnbuf[16 * 320];
  int i0 = blockIdx.x * 16, h = blockIdx.y, b = blockIdx.z;
  int lane = threadIdx.x & 31;
  const float scale = 0.125f;  // 64^-0.5
  const bf16* qbase = q + (b * 320 + i0) * 512 + h * 64;
  v16bf qa0 = frag_a(qbase, 512);
  v16bf qa1 = frag_a(qbase + 32, 512);
  const float* sb = sbias + ((b * 8 + h) * 320 + i0) * 320;     // [m][j], ld 320
  for (int jt = 0; jt < 20; ++jt) {
    const bf16* kb = k + (b * 320 + jt * 16) * 512 + h * 64;    // Bt[n=j][k=d], ld 512
    v8f acc = {};
    acc = wmma_bf16(qa0, frag_bt(kb, 512), acc);
    acc = wmma_bf16(qa1, frag_bt(kb + 32, 512), acc);
    int n = lane & 15, mb = (lane >> 4) ? 8 : 0;
    int j = jt * 16 + n;
#pragma unroll
    for (int r = 0; r < 8; ++r) {
      int m = mb + r;
      simbuf[m * 320 + j] = (acc[r] + sb[m * 320 + j]) * scale;
    }
  }
  __syncthreads();
  // softmax: lane handles row m = lane&15, half of the 320 columns
  int m = lane & 15;
  int c0 = (lane >> 4) * 160;
  float mx = -3.0e38f;
  for (int j = 0; j < 160; ++j) mx = fmaxf(mx, simbuf[m * 320 + c0 + j]);
  mx = fmaxf(mx, __shfl_xor(mx, 16, 32));
  float sum = 0.0f;
  for (int j = 0; j < 160; ++j) {
    float e = __expf(simbuf[m * 320 + c0 + j] - mx);
    simbuf[m * 320 + c0 + j] = e;
    sum += e;
  }
  sum += __shfl_xor(sum, 16, 32);
  float inv = 1.0f / sum;
  bf16* aout = attn_out + ((b * 8 + h) * 320 + i0) * 320;
  for (int j = 0; j < 160; ++j) {
    float p = simbuf[m * 320 + c0 + j] * inv;
    bf16 pb = (bf16)p;
    attnbuf[m * 320 + c0 + j] = pb;
    aout[m * 320 + c0 + j] = pb;
  }
  __syncthreads();
  // ctx[i, h*64+d] = attn[16,320] @ v[320,64]
#pragma unroll
  for (int nt = 0; nt < 4; ++nt) {
    v8f acc = {};
    for (int kt = 0; kt < 20; ++kt) {
      v16bf a  = frag_a(attnbuf + kt * 32, 320);
      v16bf bb = frag_b(v + (b * 320 + kt * 32) * 512 + h * 64 + nt * 16, 512);
      acc = wmma_bf16(a, bb, acc);
    }
    int n = lane & 15, mb = (lane >> 4) ? 8 : 0;
#pragma unroll
    for (int r = 0; r < 8; ++r)
      ctx[(b * 320 + i0 + mb + r) * 512 + h * 64 + nt * 16 + n] = acc[r];
  }
}

// ae[h,c] = sum_j attn[b,h,i,j]*edges[b,c,i,j]; ctx += ae @ W_e_block + b_e
__global__ __launch_bounds__(32) void edgeout_kernel(
    const bf16* __restrict__ attn, const bf16* __restrict__ edges,
    const float* __restrict__ We, const float* __restrict__ be,
    float* __restrict__ ctx) {
  __shared__ float ae[8 * 64];
  int i = blockIdx.x, b = blockIdx.y;
  int lane = threadIdx.x & 31;
  const bf16* Abase = attn + b * 819200 + i * 320;              // [h][j], ld 102400
  const bf16* Ebase = edges + b * 64 * 102400 + i * 320;        // Bt[n=c][k=j], ld 102400
#pragma unroll
  for (int nt = 0; nt < 4; ++nt) {
    v8f acc = {};
    for (int kt = 0; kt < 20; ++kt) {
      v16bf a  = frag_a_rows8(Abase + kt * 32, 102400);
      v16bf bb = frag_bt(Ebase + nt * 16 * 102400 + kt * 32, 102400);
      acc = wmma_bf16(a, bb, acc);
    }
    if (lane < 16) {
#pragma unroll
      for (int r = 0; r < 8; ++r) ae[r * 64 + nt * 16 + lane] = acc[r];
    }
  }
  __syncthreads();
  float* cbase = ctx + (b * 320 + i) * 512;
#pragma unroll
  for (int t = 0; t < 16; ++t) {
    int o = lane + t * 32;            // o = h*64 + d
    int h = o >> 6;
    float a2 = be[o];
    for (int c = 0; c < 64; ++c)
      a2 += ae[h * 64 + c] * We[c * 512 + o];
    cbase[o] += a2;
  }
}

extern "C" void kernel_launch(void* const* d_in, const int* in_sizes, int n_in,
                              void* d_out, int out_size, void* d_ws, size_t ws_size,
                              hipStream_t stream) {
  (void)in_sizes; (void)n_in; (void)out_size; (void)ws_size;
  const float* nodes = (const float*)d_in[0];
  const float* edges = (const float*)d_in[1];
  const float* W_exp = (const float*)d_in[2];
  const float* b_exp = (const float*)d_in[3];
  const float* W_q   = (const float*)d_in[4];
  const float* b_q   = (const float*)d_in[5];
  const float* W_kv  = (const float*)d_in[6];
  const float* b_kv  = (const float*)d_in[7];
  const float* W_e   = (const float*)d_in[8];
  const float* b_e   = (const float*)d_in[9];
  const float* W_out = (const float*)d_in[10];
  const float* b_out = (const float*)d_in[11];
  float* out = (float*)d_out;

  char* ws = (char*)d_ws;
  size_t off = 0;
  auto alloc = [&](size_t bytes) -> void* {
    void* p = ws + off;
    off = (off + bytes + 255) & ~(size_t)255;
    return p;
  };
  const int MN = 640;  // B*N
  bf16* edges_bf = (bf16*)alloc((size_t)13107200 * 2);
  bf16* nodes_bf = (bf16*)alloc((size_t)MN * 512 * 2);
  bf16* Wexp_bf  = (bf16*)alloc((size_t)512 * 512 * 2);
  bf16* Wq_bf    = (bf16*)alloc((size_t)512 * 512 * 2);
  bf16* Wkv_bf   = (bf16*)alloc((size_t)512 * 1024 * 2);
  bf16* We_bf    = (bf16*)alloc((size_t)64 * 512 * 2);
  bf16* Wout_bf  = (bf16*)alloc((size_t)512 * 512 * 2);
  bf16* x_bf     = (bf16*)alloc((size_t)MN * 512 * 2);
  bf16* q_bf     = (bf16*)alloc((size_t)MN * 512 * 2);
  bf16* k_bf     = (bf16*)alloc((size_t)MN * 512 * 2);
  bf16* v_bf     = (bf16*)alloc((size_t)MN * 512 * 2);
  bf16* qe_bf    = (bf16*)alloc((size_t)MN * 512 * 2);
  float* sbias   = (float*)alloc((size_t)2 * 8 * 320 * 320 * 4);
  bf16* attn_bf  = (bf16*)alloc((size_t)2 * 8 * 320 * 320 * 2);
  float* ctx     = (float*)alloc((size_t)MN * 512 * 4);
  bf16* ctx_bf   = (bf16*)alloc((size_t)MN * 512 * 2);

  auto conv = [&](const float* src, bf16* dst, int n) {
    int n4 = n / 4;
    f32_to_bf16_v4<<<dim3((n4 + 255) / 256), dim3(256), 0, stream>>>(
        (const v4f*)src, (v4bf*)dst, n4);
  };
  conv(edges, edges_bf, 13107200);
  conv(nodes, nodes_bf, MN * 512);
  conv(W_exp, Wexp_bf, 512 * 512);
  conv(W_q,   Wq_bf,   512 * 512);
  conv(W_kv,  Wkv_bf,  512 * 1024);
  conv(W_e,   We_bf,   64 * 512);
  conv(W_out, Wout_bf, 512 * 512);

  dim3 gg(512 / 64, MN / 32);
  // x = nodes @ W_exp + b_exp
  gemm_bf16_kernel<false><<<gg, 256, 0, stream>>>(nodes_bf, 512, Wexp_bf, 512, 0, b_exp,
                                                  x_bf, 512, MN, 512, 512);
  // q = x @ W_q + b_q
  gemm_bf16_kernel<false><<<gg, 256, 0, stream>>>(x_bf, 512, Wq_bf, 512, 0, b_q,
                                                  q_bf, 512, MN, 512, 512);
  // k, v = split(x @ W_kv + b_kv)
  gemm_bf16_kernel<false><<<gg, 256, 0, stream>>>(x_bf, 512, Wkv_bf, 1024, 0, b_kv,
                                                  k_bf, 512, MN, 512, 512);
  gemm_bf16_kernel<false><<<gg, 256, 0, stream>>>(x_bf, 512, Wkv_bf, 1024, 512, b_kv,
                                                  v_bf, 512, MN, 512, 512);
  // qe = per-head q @ W_e_block^T
  qe_kernel<<<dim3(MN / 16, 8), 32, 0, stream>>>(q_bf, We_bf, qe_bf);
  // sbias = qe . edges
  sbias_kernel<<<dim3(320, 2), 128, 0, stream>>>(qe_bf, edges_bf, sbias);
  // attention core
  attn_kernel<<<dim3(20, 8, 2), 32, 0, stream>>>(q_bf, k_bf, v_bf, sbias, attn_bf, ctx);
  // edge value contribution (+ b_e, since attn rows sum to 1)
  edgeout_kernel<<<dim3(320, 2), 32, 0, stream>>>(attn_bf, edges_bf, W_e, b_e, ctx);
  // out = ctx @ W_out + b_out
  conv(ctx, ctx_bf, MN * 512);
  gemm_bf16_kernel<true><<<gg, 256, 0, stream>>>(ctx_bf, 512, Wout_bf, 512, 0, b_out,
                                                 out, 512, MN, 512, 512);
}
